// GraphTransformerEncoder_35021163331783
// MI455X (gfx1250) — compile-verified, matrix operand order checked by reading, not run on hardware
//
#include <hip/hip_runtime.h>
#include <math.h>

#define D 128
#define H 8

// ---------- CDNA5 WMMA vector types ----------
typedef __attribute__((ext_vector_type(16))) __bf16 v16bf;
typedef __attribute__((ext_vector_type(8)))  float  v8f;
typedef __attribute__((ext_vector_type(4)))  unsigned int u32x4;

// ---------- helpers ----------
__device__ __forceinline__ unsigned short f2bf(float f) {
  unsigned int b = __float_as_uint(f);
  b += 0x7FFFu + ((b >> 16) & 1u);        // round-to-nearest-even
  return (unsigned short)(b >> 16);
}
// monotone f32 <-> ordered-u32 map (for atomic segment max)
__device__ __forceinline__ unsigned enc_f32(float f) {
  unsigned b = __float_as_uint(f);
  return (b & 0x80000000u) ? ~b : (b | 0x80000000u);
}
__device__ __forceinline__ float dec_f32(unsigned k) {
  unsigned b = (k & 0x80000000u) ? (k & 0x7FFFFFFFu) : ~k;
  return __uint_as_float(b);
}

// ---------- prep kernels ----------
// transpose one 128x128 f32 W[k][n] -> bf16 Wt[n][k] (B-fragment loads contiguous in K)
__global__ void k_wt(const float* __restrict__ W, unsigned short* __restrict__ Wt) {
  int i = blockIdx.x * 256 + threadIdx.x;          // < 16384
  int nn = i >> 7, kk = i & 127;
  Wt[nn * D + kk] = f2bf(W[kk * D + nn]);
}

__global__ void k_f32_to_bf16(const float* __restrict__ in, unsigned short* __restrict__ out, int count) {
  int i = blockIdx.x * blockDim.x + threadIdx.x;
  if (i < count) out[i] = f2bf(in[i]);
}

// gather node embeddings into x (f32, in d_out) and xh (bf16 shadow for WMMA)
__global__ void k_gather_x(const int* __restrict__ ids, const float* __restrict__ emb,
                           float* __restrict__ x, unsigned short* __restrict__ xh, int n) {
  int i = blockIdx.x * blockDim.x + threadIdx.x;
  if (i >= n * D) return;
  int node = i >> 7, c = i & 127;
  float v = emb[(size_t)ids[node] * D + c];
  x[i]  = v;
  xh[i] = f2bf(v);
}

// ---------- WMMA GEMM: C[M x 128] = bf16(A[M x 128]) @ bf16(Wt^T) + bias ----------
// One wave handles 16 rows x ALL 128 output columns: 8 accumulators, 32 WMMAs.
// Per K-step: 1 A-fragment load + 8 distinct B-fragment loads (one clause), then
// 8 back-to-back WMMAs. First WMMA of each chain uses inline-0 C; bias is added
// in the epilogue.  grid.x = ceil(M/16) row tiles, block = 1 wave (32 lanes).
__global__ void __launch_bounds__(32) k_gemm(const unsigned short* __restrict__ A,
                                             const unsigned short* __restrict__ Wt,
                                             const float* __restrict__ bias,
                                             float* __restrict__ C, int M) {
  int rt = blockIdx.x;
  int lane = threadIdx.x;
  int l15 = lane & 15, hi = lane >> 4;
  bool full = (rt * 16 + 16 <= M);                 // exact-tile fast path (always true for nodes)
  int arow = rt * 16 + l15; if (arow >= M) arow = M - 1;
  const unsigned short* a0 = A  + (size_t)arow * D + hi * 8;
  const unsigned short* b0 = Wt + (size_t)l15 * D + hi * 8;   // ct advances by 16 rows = 16*D elems

  v8f acc[8];
#pragma unroll
  for (int ct = 0; ct < 8; ++ct) acc[ct] = (v8f){};           // inline-0 C on first WMMA

  union Frag { u32x4 u[2]; v16bf v; };
#pragma unroll
  for (int k = 0; k < D; k += 32) {
    Frag fa;
    fa.u[0] = *reinterpret_cast<const u32x4*>(a0 + k);
    fa.u[1] = *reinterpret_cast<const u32x4*>(a0 + k + 16);
    Frag fb[8];
#pragma unroll
    for (int ct = 0; ct < 8; ++ct) {                          // one big load clause
      fb[ct].u[0] = *reinterpret_cast<const u32x4*>(b0 + ct * 16 * D + k);
      fb[ct].u[1] = *reinterpret_cast<const u32x4*>(b0 + ct * 16 * D + k + 16);
    }
#pragma unroll
    for (int ct = 0; ct < 8; ++ct)                            // 8 back-to-back WMMAs
      acc[ct] = __builtin_amdgcn_wmma_f32_16x16x32_bf16(false, fa.v, false, fb[ct].v,
                                                        (short)0, acc[ct], false, false);
  }

  int m0 = rt * 16 + hi * 8;
  if (full) {
#pragma unroll
    for (int ct = 0; ct < 8; ++ct) {
      int n = ct * 16 + l15;
      float bv = bias[n];
#pragma unroll
      for (int r = 0; r < 8; ++r)
        C[(size_t)(m0 + r) * D + n] = acc[ct][r] + bv;        // lanes 0-15 contiguous -> coalesced
    }
  } else {
#pragma unroll
    for (int ct = 0; ct < 8; ++ct) {
      int n = ct * 16 + l15;
      float bv = bias[n];
#pragma unroll
      for (int r = 0; r < 8; ++r)
        if (m0 + r < M) C[(size_t)(m0 + r) * D + n] = acc[ct][r] + bv;
    }
  }
}

// ---------- segment softmax / message passing ----------
__global__ void k_init_seg(unsigned* __restrict__ amaxk, float* __restrict__ denom, int count) {
  int i = blockIdx.x * blockDim.x + threadIdx.x;
  if (i < count) { amaxk[i] = 0u; denom[i] = 0.0f; }
}

__global__ void k_edge_alpha(const int* __restrict__ ei, const int* __restrict__ et,
                             const float* __restrict__ Q, const float* __restrict__ K,
                             const float* __restrict__ RP, float* __restrict__ alpha,
                             unsigned* __restrict__ amaxk, int E_) {
  int i = blockIdx.x * blockDim.x + threadIdx.x;
  if (i >= E_ * H) return;
  int e = i >> 3, h = i & 7;
  int src = ei[e], dst = ei[E_ + e], typ = et[e];
  const float4* q = reinterpret_cast<const float4*>(Q  + (size_t)dst * D + h * 16);
  const float4* k = reinterpret_cast<const float4*>(K  + (size_t)src * D + h * 16);
  const float4* r = reinterpret_cast<const float4*>(RP + (size_t)typ * D + h * 16);
  float s = 0.0f;
#pragma unroll
  for (int j = 0; j < 4; ++j) {
    float4 qa = q[j], ka = k[j], ra = r[j];
    s += qa.x * (ka.x + ra.x) + qa.y * (ka.y + ra.y) +
         qa.z * (ka.z + ra.z) + qa.w * (ka.w + ra.w);
  }
  s *= 0.25f;                      // 1/sqrt(C), C=16
  alpha[i] = s;
  atomicMax(&amaxk[dst * H + h], enc_f32(s));
}

__global__ void k_edge_exp(const int* __restrict__ ei, float* __restrict__ alpha,
                           const unsigned* __restrict__ amaxk, float* __restrict__ denom, int E_) {
  int i = blockIdx.x * blockDim.x + threadIdx.x;
  if (i >= E_ * H) return;
  int e = i >> 3, h = i & 7;
  int dst = ei[E_ + e];
  float am = dec_f32(amaxk[dst * H + h]);
  float ex = __expf(alpha[i] - am);
  alpha[i] = ex;
  atomicAdd(&denom[dst * H + h], ex);
}

// one block (128 threads) per edge: scatter-add message into accum (pre-loaded with skip term)
__global__ void k_edge_scatter(const int* __restrict__ ei, const int* __restrict__ et,
                               const float* __restrict__ V, const float* __restrict__ RP,
                               const float* __restrict__ exv, const float* __restrict__ denom,
                               float* __restrict__ accum, int E_) {
  int e = blockIdx.x, c = threadIdx.x, h = c >> 4;
  int src = ei[e], dst = ei[E_ + e], typ = et[e];
  float coef = exv[(size_t)e * H + h] / (denom[dst * H + h] + 1e-16f);
  float val = (V[(size_t)src * D + c] + RP[(size_t)typ * D + c]) * coef;
  atomicAdd(&accum[(size_t)dst * D + c], val);
}

// one wave32 per node: LayerNorm -> ReLU -> residual into x; also refresh bf16 shadow xh
__global__ void k_norm_residual(const float* __restrict__ accum, float* __restrict__ x,
                                unsigned short* __restrict__ xh,
                                const float* __restrict__ g, const float* __restrict__ b, int n) {
  int wave = threadIdx.x >> 5, lane = threadIdx.x & 31;
  int node = blockIdx.x * 8 + wave;
  if (node >= n) return;
  float4 y = reinterpret_cast<const float4*>(accum + (size_t)node * D)[lane];
  float s = y.x + y.y + y.z + y.w;
#pragma unroll
  for (int m = 16; m >= 1; m >>= 1) s += __shfl_xor(s, m);
  float mu = s * (1.0f / D);
  float d0 = y.x - mu, d1 = y.y - mu, d2 = y.z - mu, d3 = y.w - mu;
  float vs = d0 * d0 + d1 * d1 + d2 * d2 + d3 * d3;
#pragma unroll
  for (int m = 16; m >= 1; m >>= 1) vs += __shfl_xor(vs, m);
  float inv = rsqrtf(vs * (1.0f / D) + 1e-5f);
  int c0 = lane * 4;
  float4 gg = *reinterpret_cast<const float4*>(g + c0);
  float4 bb = *reinterpret_cast<const float4*>(b + c0);
  float4* xv = reinterpret_cast<float4*>(x + (size_t)node * D);
  float4 xo = xv[lane];
  xo.x += fmaxf(d0 * inv * gg.x + bb.x, 0.0f);
  xo.y += fmaxf(d1 * inv * gg.y + bb.y, 0.0f);
  xo.z += fmaxf(d2 * inv * gg.z + bb.z, 0.0f);
  xo.w += fmaxf(d3 * inv * gg.w + bb.w, 0.0f);
  xv[lane] = xo;
  ushort4 hsh;
  hsh.x = f2bf(xo.x); hsh.y = f2bf(xo.y); hsh.z = f2bf(xo.z); hsh.w = f2bf(xo.w);
  reinterpret_cast<ushort4*>(xh + (size_t)node * D)[lane] = hsh;
}

// ---------- graph embedding (mean over nodes) ----------
__global__ void k_zero_gsum(float* __restrict__ gsum) { gsum[threadIdx.x] = 0.0f; }

__global__ void k_colsum(const float* __restrict__ x, float* __restrict__ gsum, int n) {
  int c = threadIdx.x;
  float s = 0.0f;
  for (int node = blockIdx.x; node < n; node += gridDim.x) s += x[(size_t)node * D + c];
  atomicAdd(&gsum[c], s);
}

__global__ void k_finalize(const float* __restrict__ gsum, float* __restrict__ out, int n) {
  int c = threadIdx.x;
  out[c] = gsum[c] * (1.0f / (float)n);
}

// ---------- driver ----------
extern "C" void kernel_launch(void* const* d_in, const int* in_sizes, int n_in,
                              void* d_out, int out_size, void* d_ws, size_t ws_size,
                              hipStream_t stream) {
  const int*   node_ids   = (const int*)d_in[0];
  const int*   edge_index = (const int*)d_in[1];
  const int*   edge_type  = (const int*)d_in[2];
  const float* ent_emb    = (const float*)d_in[3];
  const float* rel_emb    = (const float*)d_in[4];
  // weight order used internally: q,k,v,s,e
  const float* W[5] = { (const float*)d_in[5], (const float*)d_in[7], (const float*)d_in[9],
                        (const float*)d_in[13], (const float*)d_in[11] };
  const float* B[5] = { (const float*)d_in[6], (const float*)d_in[8], (const float*)d_in[10],
                        (const float*)d_in[14], (const float*)d_in[12] };
  const float* ln_g = (const float*)d_in[15];
  const float* ln_b = (const float*)d_in[16];

  const int n    = in_sizes[0];                 // 100000 nodes
  const int e    = in_sizes[2];                 // 600000 edges
  const int nrel = in_sizes[4] / D;             // 1000 relations
  const int L    = in_sizes[5] / (D * D);       // 3 layers

  float* x = (float*)d_out;                     // x lives in d_out[0 .. n*D)

  // ---- workspace carve (bump allocator, 256B aligned) ----
  char* p = (char*)d_ws;
  auto carve = [&](size_t bytes) -> char* {
    char* r = p; p += (bytes + 255) & ~(size_t)255; return r;
  };
  unsigned short* xh    = (unsigned short*)carve((size_t)n * D * 2);
  unsigned short* relh  = (unsigned short*)carve((size_t)nrel * D * 2);
  unsigned short* Wt    = (unsigned short*)carve((size_t)L * 5 * D * D * 2);
  float*    RP    = (float*)carve((size_t)nrel * D * 4);
  float*    Qb    = (float*)carve((size_t)n * D * 4);
  float*    Kb    = (float*)carve((size_t)n * D * 4);
  float*    Vb    = (float*)carve((size_t)n * D * 4);
  float*    accum = (float*)carve((size_t)n * D * 4);
  float*    alpha = (float*)carve((size_t)e * H * 4);
  unsigned* amaxk = (unsigned*)carve((size_t)n * H * 4);
  float*    denom = (float*)carve((size_t)n * H * 4);
  float*    gsum  = (float*)carve(D * 4);

  // ---- one-time (per call) prep ----
  for (int l = 0; l < L; ++l)
    for (int w = 0; w < 5; ++w)
      k_wt<<<64, 256, 0, stream>>>(W[w] + (size_t)l * D * D, Wt + ((size_t)l * 5 + w) * D * D);
  k_f32_to_bf16<<<(nrel * D + 255) / 256, 256, 0, stream>>>(rel_emb, relh, nrel * D);
  k_gather_x<<<(n * D + 255) / 256, 256, 0, stream>>>(node_ids, ent_emb, x, xh, n);

  // ---- layers ----
  for (int l = 0; l < L; ++l) {
    dim3 gq((n + 15) / 16);
    k_gemm<<<gq, 32, 0, stream>>>(xh, Wt + ((size_t)l * 5 + 0) * D * D, B[0] + l * D, Qb, n);
    k_gemm<<<gq, 32, 0, stream>>>(xh, Wt + ((size_t)l * 5 + 1) * D * D, B[1] + l * D, Kb, n);
    k_gemm<<<gq, 32, 0, stream>>>(xh, Wt + ((size_t)l * 5 + 2) * D * D, B[2] + l * D, Vb, n);
    // skip path (x@Ws + bs) written straight into accum; edge messages atomically add on top
    k_gemm<<<gq, 32, 0, stream>>>(xh, Wt + ((size_t)l * 5 + 3) * D * D, B[3] + l * D, accum, n);
    dim3 gr((nrel + 15) / 16);
    k_gemm<<<gr, 32, 0, stream>>>(relh, Wt + ((size_t)l * 5 + 4) * D * D, B[4] + l * D, RP, nrel);

    k_init_seg<<<(n * H + 255) / 256, 256, 0, stream>>>(amaxk, denom, n * H);
    k_edge_alpha<<<(e * H + 255) / 256, 256, 0, stream>>>(edge_index, edge_type, Qb, Kb, RP,
                                                          alpha, amaxk, e);
    k_edge_exp<<<(e * H + 255) / 256, 256, 0, stream>>>(edge_index, alpha, amaxk, denom, e);
    k_edge_scatter<<<e, D, 0, stream>>>(edge_index, edge_type, Vb, RP, alpha, denom, accum, e);
    k_norm_residual<<<(n + 7) / 8, 256, 0, stream>>>(accum, x, xh, ln_g + l * D, ln_b + l * D, n);
  }

  // ---- graph embedding ----
  k_zero_gsum<<<1, D, 0, stream>>>(gsum);
  k_colsum<<<256, D, 0, stream>>>(x, gsum, n);
  k_finalize<<<1, D, 0, stream>>>(gsum, x + (size_t)n * D, n);
}
